// Model_Recursive_LSTM_v2_26104811225245
// MI455X (gfx1250) — compile-verified, matrix operand order checked by reading, not run on hardware
//
#include <hip/hip_runtime.h>
#include <hip/hip_bf16.h>

// ---------------------------------------------------------------------------
// CDNA5 (gfx1250) WMMA building blocks. wave32; 16x16x32 f16 -> f32 accum.
// Fragment index maps per CDNA5 ISA 7.12.2 (16-bit A 16x32, B 32x16, f32 C/D).
// ---------------------------------------------------------------------------
typedef __attribute__((ext_vector_type(16))) _Float16 v16h;
typedef __attribute__((ext_vector_type(8)))  float    v8f;

#if __has_builtin(__builtin_amdgcn_global_load_async_to_lds_b128) && \
    __has_builtin(__builtin_amdgcn_global_load_async_to_lds_b32) &&  \
    __has_builtin(__builtin_amdgcn_s_wait_asynccnt)
#define USE_ASYNC_LDS 1
#else
#define USE_ASYNC_LDS 0
#endif

// Parameter types per hipcc diagnostics: generic 'int*' (b32) and generic
// pointer to 'int __attribute__((vector_size(16)))' (b128); no address-space
// qualifier on the builtin parameters.
typedef int async_v4i __attribute__((vector_size(16)));

__device__ __forceinline__ void async_b128(const void* g, void* l) {
#if USE_ASYNC_LDS
  __builtin_amdgcn_global_load_async_to_lds_b128((async_v4i*)g, (async_v4i*)l, 0, 0);
#endif
}
__device__ __forceinline__ void async_b32(const void* g, void* l) {
#if USE_ASYNC_LDS
  __builtin_amdgcn_global_load_async_to_lds_b32((int*)g, (int*)l, 0, 0);
#endif
}
__device__ __forceinline__ void async_wait() {
#if USE_ASYNC_LDS
  __builtin_amdgcn_s_wait_asynccnt(0);
#endif
}

__device__ __forceinline__ v8f wmma_f16(v16h a, v16h b, v8f c) {
  return __builtin_amdgcn_wmma_f32_16x16x32_f16(false, a, false, b, (short)0, c, false, false);
}

// K index of fragment element e for this lane (lanes 0-15: K base 0, 16-31: base 8)
__device__ __forceinline__ int frag_k(int e, int lane) {
  int kb = (lane >> 4) << 3;
  return (e < 8) ? (kb + e) : (16 + kb + (e - 8));
}

// A-tile from fp16 row-major LDS
__device__ __forceinline__ v16h load_A_lds(const _Float16* As, int lda, int k0, int lane) {
  int m = lane & 15;
  v16h a;
#pragma unroll
  for (int e = 0; e < 16; ++e) a[e] = As[m * lda + k0 + frag_k(e, lane)];
  return a;
}

// A-tile from fp32 row-major LDS (async-staged), convert on fragment build
__device__ __forceinline__ v16h load_A_lds_f32(const float* As, int lda, int k0, int lane) {
  int m = lane & 15;
  v16h a;
#pragma unroll
  for (int e = 0; e < 16; ++e) a[e] = (_Float16)As[m * lda + k0 + frag_k(e, lane)];
  return a;
}

// B-tile from W^T layout [N][ldk] fp16 in LDS (B[k][n] = W[n][k])
__device__ __forceinline__ v16h load_B_lds(const _Float16* Wt, int ldk, int n0, int k0, int lane) {
  int n = n0 + (lane & 15);
  v16h b;
#pragma unroll
  for (int e = 0; e < 16; ++e) b[e] = Wt[n * ldk + k0 + frag_k(e, lane)];
  return b;
}

// B-tile from global fp32 weights [N][K], bounds-checked, fp32->fp16 convert
__device__ __forceinline__ v16h load_B_glb(const float* __restrict__ W, int N, int K,
                                           int n0, int k0, int lane) {
  int n = n0 + (lane & 15);
  v16h b;
#pragma unroll
  for (int e = 0; e < 16; ++e) {
    int k = k0 + frag_k(e, lane);
    b[e] = (_Float16)((n < N && k < K) ? W[(size_t)n * K + k] : 0.f);
  }
  return b;
}

__device__ __forceinline__ float sigmf(float x) { return 1.f / (1.f + __expf(-x)); }
__device__ __forceinline__ float actf(float x, int act) {
  if (act == 1) return x > 0.f ? x : (__expf(x) - 1.f);   // ELU
  if (act == 2) return x > 0.f ? x : 0.01f * x;           // LeakyReLU(0.01)
  return x;
}

// ---------------------------------------------------------------------------
// Fused LSTM, H=100 (expr_lstm / vec_lstm_f / vec_lstm_b).
// Weights LDS-resident (fp16) for the whole time loop: Wi[400x32], Wh[400x128].
// One block = 16 sequences; 8 waves cover the 25 gate N-tiles.
// x_t tile async-DMA'd into LDS each step (pad region pre-zeroed once).
// ---------------------------------------------------------------------------
__global__ void lstm100_kernel(const float* __restrict__ X, int T, int I, int reverse,
                               const float* __restrict__ Wi, const float* __restrict__ Wh,
                               const float* __restrict__ bi, const float* __restrict__ bh,
                               float* __restrict__ outH, int outStride, int outOff) {
  extern __shared__ char smem[];
  _Float16* WiH = (_Float16*)smem;             // 400*32
  _Float16* WhH = WiH + 400 * 32;              // 400*128
  float*    bias = (float*)(WhH + 400 * 128);  // 400
  float*    Axf = bias + 400;                  // 16*32 f32
  _Float16* Ah  = (_Float16*)(Axf + 16 * 32);  // 16*128
  float*    G   = (float*)(Ah + 16 * 128);     // 16*400 gate accum
  float*    Cs  = G + 16 * 400;                // 16*100
  float*    Hs  = Cs + 16 * 100;               // 16*100

  const int tid = threadIdx.x, lane = tid & 31, wave = tid >> 5;
  const int seq0 = blockIdx.x * 16;

  for (int i = tid; i < 400 * 32; i += 256) {
    int n = i >> 5, k = i & 31;
    WiH[i] = (_Float16)((k < I) ? Wi[(size_t)n * I + k] : 0.f);
  }
  for (int i = tid; i < 400 * 128; i += 256) {
    int n = i >> 7, k = i & 127;
    WhH[i] = (_Float16)((k < 100) ? Wh[(size_t)n * 100 + k] : 0.f);
  }
  for (int i = tid; i < 400; i += 256) bias[i] = bi[i] + bh[i];
  for (int i = tid; i < 16 * 32; i += 256) Axf[i] = 0.f;
  for (int i = tid; i < 16 * 128; i += 256) Ah[i] = (_Float16)0.f;
  for (int i = tid; i < 16 * 100; i += 256) { Cs[i] = 0.f; Hs[i] = 0.f; }
  __syncthreads();

  for (int ts = 0; ts < T; ++ts) {
    const int st = reverse ? (T - 1 - ts) : ts;
#if USE_ASYNC_LDS
    for (int i = tid; i < 16 * I; i += 256) {
      int s = i / I, k = i % I;
      async_b32(X + ((size_t)(seq0 + s) * T + st) * I + k, Axf + s * 32 + k);
    }
    async_wait();
#else
    for (int i = tid; i < 16 * 32; i += 256) {
      int s = i >> 5, k = i & 31;
      Axf[i] = (k < I) ? X[((size_t)(seq0 + s) * T + st) * I + k] : 0.f;
    }
#endif
    __syncthreads();

    for (int t = wave; t < 25; t += 8) {
      v8f acc = {};
      acc = wmma_f16(load_A_lds_f32(Axf, 32, 0, lane), load_B_lds(WiH, 32, t * 16, 0, lane), acc);
#pragma unroll
      for (int k0 = 0; k0 < 128; k0 += 32)
        acc = wmma_f16(load_A_lds(Ah, 128, k0, lane), load_B_lds(WhH, 128, t * 16, k0, lane), acc);
      const int nn = (lane & 15) + t * 16;
      const int mb = (lane >> 4) << 3;
#pragma unroll
      for (int r = 0; r < 8; ++r) G[(mb + r) * 400 + nn] = acc[r];
    }
    __syncthreads();

    for (int e = tid; e < 16 * 100; e += 256) {
      int s = e / 100, j = e % 100;
      float ig = G[s * 400 + j]       + bias[j];
      float fg = G[s * 400 + 100 + j] + bias[100 + j];
      float gg = G[s * 400 + 200 + j] + bias[200 + j];
      float og = G[s * 400 + 300 + j] + bias[300 + j];
      float c = sigmf(fg) * Cs[e] + sigmf(ig) * tanhf(gg);
      float h = sigmf(og) * tanhf(c);
      Cs[e] = c; Hs[e] = h;
      Ah[s * 128 + j] = (_Float16)h;
    }
    __syncthreads();
  }

  for (int e = tid; e < 16 * 100; e += 256) {
    int s = e / 100, j = e % 100;
    outH[(size_t)(seq0 + s) * outStride + outOff + j] = Hs[e];
  }
}

// ---------------------------------------------------------------------------
// LSTM with E=180 in/hidden, sequence length 2 (comps/nodes/roots LSTMs).
// x_t selected via (batchStride, selStride, idx[t]); strides are multiples of 4
// floats so the 16x180 x-tile is staged with async b128 DMA (45 quads/row).
// Weights streamed from global (fp16 Wi+Wh would be 518KB > 320KB LDS).
// ---------------------------------------------------------------------------
__global__ void lstm180_seq2_kernel(const float* __restrict__ X, int xBatch, int xSel,
                                    int i0, int i1,
                                    const float* __restrict__ Wi, const float* __restrict__ Wh,
                                    const float* __restrict__ bi, const float* __restrict__ bh,
                                    float* __restrict__ outH) {
  extern __shared__ char smem[];
  float*    Axf = (float*)smem;                 // 16*192 f32
  _Float16* Ah  = (_Float16*)(Axf + 16 * 192);  // 16*192
  float*    G   = (float*)(Ah + 16 * 192);      // 16*720
  float*    Cs  = G + 16 * 720;                 // 16*180
  float*    Hs  = Cs + 16 * 180;                // 16*180

  const int tid = threadIdx.x, lane = tid & 31, wave = tid >> 5;
  const int r0 = blockIdx.x * 16;
  for (int i = tid; i < 16 * 192; i += 256) { Axf[i] = 0.f; Ah[i] = (_Float16)0.f; }
  for (int i = tid; i < 16 * 180; i += 256) { Cs[i] = 0.f; Hs[i] = 0.f; }
  __syncthreads();

  const int sel[2] = {i0, i1};
  for (int st = 0; st < 2; ++st) {
#if USE_ASYNC_LDS
    for (int q = tid; q < 16 * 45; q += 256) {   // 45 x b128 per row covers 180 floats
      int s = q / 45, kq = (q % 45) << 2;
      async_b128(X + (size_t)(r0 + s) * xBatch + (size_t)sel[st] * xSel + kq,
                 Axf + s * 192 + kq);
    }
    async_wait();
#else
    for (int i = tid; i < 16 * 192; i += 256) {
      int s = i / 192, k = i % 192;
      if (k < 180)
        Axf[i] = X[(size_t)(r0 + s) * xBatch + (size_t)sel[st] * xSel + k];
    }
#endif
    __syncthreads();

    for (int t = wave; t < 45; t += 8) {
      v8f acc = {};
#pragma unroll
      for (int k0 = 0; k0 < 192; k0 += 32)
        acc = wmma_f16(load_A_lds_f32(Axf, 192, k0, lane),
                       load_B_glb(Wi, 720, 180, t * 16, k0, lane), acc);
#pragma unroll
      for (int k0 = 0; k0 < 192; k0 += 32)
        acc = wmma_f16(load_A_lds(Ah, 192, k0, lane),
                       load_B_glb(Wh, 720, 180, t * 16, k0, lane), acc);
      const int nn = (lane & 15) + t * 16;
      const int mb = (lane >> 4) << 3;
#pragma unroll
      for (int r = 0; r < 8; ++r) G[(mb + r) * 720 + nn] = acc[r];
    }
    __syncthreads();

    for (int e = tid; e < 16 * 180; e += 256) {
      int s = e / 180, j = e % 180;
      float ig = G[s * 720 + j]       + bi[j]       + bh[j];
      float fg = G[s * 720 + 180 + j] + bi[180 + j] + bh[180 + j];
      float gg = G[s * 720 + 360 + j] + bi[360 + j] + bh[360 + j];
      float og = G[s * 720 + 540 + j] + bi[540 + j] + bh[540 + j];
      float c = sigmf(fg) * Cs[e] + sigmf(ig) * tanhf(gg);
      float h = sigmf(og) * tanhf(c);
      Cs[e] = c; Hs[e] = h;
      Ah[s * 192 + j] = (_Float16)h;
    }
    __syncthreads();
  }

  for (int e = tid; e < 16 * 180; e += 256) {
    int s = e / 180, j = e % 180;
    outH[(size_t)(r0 + s) * 180 + j] = Hs[e];
  }
}

// ---------------------------------------------------------------------------
// Generic WMMA GEMM: C = act(A[MxK] * W^T + bias), W is [NxK] fp32 (torch layout).
// Block: 8 waves -> 16 x 128 output tile. Per 32-wide K chunk:
//   A 16x32 f32  -> LDS (async b32 DMA when the tile is fully in range),
//   B 128x32 f16 -> LDS cooperatively (coalesced along K), fragments via ds_load.
// ---------------------------------------------------------------------------
__global__ void gemm_kernel(const float* __restrict__ A, int lda,
                            const float* __restrict__ W, const float* __restrict__ bias,
                            float* __restrict__ C, int ldc, int M, int N, int K, int act) {
  extern __shared__ char smem[];
  float*    Asf = (float*)smem;                 // 16*32 f32
  _Float16* Bs  = (_Float16*)(Asf + 16 * 32);   // 128*32 f16
  const int tid = threadIdx.x, lane = tid & 31, wave = tid >> 5;
  const int m0 = blockIdx.x * 16;
  const int nblk = blockIdx.y * 128;
  v8f acc = {};
  for (int k0 = 0; k0 < K; k0 += 32) {
#if USE_ASYNC_LDS
    if ((m0 + 16 <= M) && (k0 + 32 <= K)) {
      for (int i = tid; i < 16 * 32; i += 256) {
        int s = i >> 5, k = i & 31;
        async_b32(A + (size_t)(m0 + s) * lda + k0 + k, Asf + i);
      }
    } else
#endif
    {
      for (int i = tid; i < 16 * 32; i += 256) {
        int s = i >> 5, k = i & 31;
        int row = m0 + s, col = k0 + k;
        Asf[i] = (row < M && col < K) ? A[(size_t)row * lda + col] : 0.f;
      }
    }
    for (int i = tid; i < 128 * 32; i += 256) {
      int n = i >> 5, k = i & 31;
      int gn = nblk + n, gk = k0 + k;
      Bs[i] = (_Float16)((gn < N && gk < K) ? W[(size_t)gn * K + gk] : 0.f);
    }
    async_wait();
    __syncthreads();
    acc = wmma_f16(load_A_lds_f32(Asf, 32, 0, lane), load_B_lds(Bs, 32, wave * 16, 0, lane), acc);
    __syncthreads();
  }
  const int nn = nblk + wave * 16 + (lane & 15);
  const int mb = m0 + ((lane >> 4) << 3);
#pragma unroll
  for (int r = 0; r < 8; ++r) {
    int m = mb + r;
    if (m < M && nn < N) C[(size_t)m * ldc + nn] = actf(acc[r] + bias[nn], act);
  }
}

// ---------------------------------------------------------------------------
// Concat kernels
// ---------------------------------------------------------------------------
__global__ void concat_comp_kernel(const float* __restrict__ first, const float* __restrict__ prog,
                                   const float* __restrict__ third, const float* __restrict__ expr,
                                   float* __restrict__ X) {
  const size_t total = (size_t)16384 * 946;
  for (size_t i = (size_t)blockIdx.x * blockDim.x + threadIdx.x; i < total;
       i += (size_t)gridDim.x * blockDim.x) {
    size_t row = i / 946; int c = (int)(i % 946);
    float v;
    if (c < 246)      v = first[row * 246 + c];
    else if (c < 446) v = prog[row * 200 + (c - 246)];
    else if (c < 846) v = third[row * 400 + (c - 446)];
    else              v = expr[row * 100 + (c - 846)];
    X[i] = v;
  }
}

__global__ void node_concat_kernel(float* __restrict__ out,
                                   const float* __restrict__ nodesH, int nodesBcast,
                                   const float* __restrict__ compsH, int compsBcast,
                                   const float* __restrict__ loops, int loopIdx) {
  const size_t total = (size_t)1024 * 395;
  for (size_t i = (size_t)blockIdx.x * blockDim.x + threadIdx.x; i < total;
       i += (size_t)gridDim.x * blockDim.x) {
    size_t row = i / 395; int c = (int)(i % 395);
    float v;
    if (c < 180)      v = nodesBcast ? nodesH[c] : nodesH[row * 180 + c];
    else if (c < 360) v = compsBcast ? compsH[c - 180] : compsH[row * 180 + (c - 180)];
    else              v = loops[row * 14 * 35 + (size_t)loopIdx * 35 + (c - 360)];
    out[i] = v;
  }
}

// ---------------------------------------------------------------------------
// Host orchestration. Params follow the data tensors as the jax pytree flat
// order: dict keys alphabetical, tuples/lists in order.
// ---------------------------------------------------------------------------
extern "C" void kernel_launch(void* const* d_in, const int* in_sizes, int n_in,
                              void* d_out, int out_size, void* d_ws, size_t ws_size,
                              hipStream_t stream) {
  (void)in_sizes; (void)n_in; (void)out_size; (void)ws_size;
  const float* in_first = (const float*)d_in[0];   // [1024,16,246]
  const float* in_vecs  = (const float*)d_in[1];   // [16384,16,8]
  const float* in_third = (const float*)d_in[2];   // [1024,16,400]
  const float* in_loops = (const float*)d_in[3];   // [1024,14,35]
  const float* in_expr  = (const float*)d_in[4];   // [16384,64,11]
  const float* cmpW[4] = {(const float*)d_in[5], (const float*)d_in[7],
                          (const float*)d_in[9], (const float*)d_in[11]};
  const float* cmpB[4] = {(const float*)d_in[6], (const float*)d_in[8],
                          (const float*)d_in[10], (const float*)d_in[12]};
  const float *compsWi = (const float*)d_in[13], *compsWh = (const float*)d_in[14],
              *compsBi = (const float*)d_in[15], *compsBh = (const float*)d_in[16];
  const float *catW0 = (const float*)d_in[17], *catB0 = (const float*)d_in[18],
              *catW1 = (const float*)d_in[19], *catB1 = (const float*)d_in[20];
  const float *encW = (const float*)d_in[21], *encB = (const float*)d_in[22];
  const float *exprWi = (const float*)d_in[23], *exprWh = (const float*)d_in[24],
              *exprBi = (const float*)d_in[25], *exprBh = (const float*)d_in[26];
  const float *noComps = (const float*)d_in[27], *noNodes = (const float*)d_in[28];
  const float *nodesWi = (const float*)d_in[29], *nodesWh = (const float*)d_in[30],
              *nodesBi = (const float*)d_in[31], *nodesBh = (const float*)d_in[32];
  const float *predW = (const float*)d_in[33], *predB = (const float*)d_in[34];
  const float *regW0 = (const float*)d_in[35], *regB0 = (const float*)d_in[36],
              *regW1 = (const float*)d_in[37], *regB1 = (const float*)d_in[38];
  const float *rootsWi = (const float*)d_in[39], *rootsWh = (const float*)d_in[40],
              *rootsBi = (const float*)d_in[41], *rootsBh = (const float*)d_in[42];
  const float *vbWi = (const float*)d_in[43], *vbWh = (const float*)d_in[44],
              *vbBi = (const float*)d_in[45], *vbBh = (const float*)d_in[46];
  const float *vfWi = (const float*)d_in[47], *vfWh = (const float*)d_in[48],
              *vfBi = (const float*)d_in[49], *vfBh = (const float*)d_in[50];

  float* ws = (float*)d_ws;
  size_t o = 0;
  float* w_vec  = ws + o; o += (size_t)16384 * 16 * 8;
  float* w_prog = ws + o; o += (size_t)16384 * 200;
  float* w_expr = ws + o; o += (size_t)16384 * 100;
  float* w_xcat = ws + o; o += (size_t)16384 * 946;
  float* w_y1   = ws + o; o += (size_t)16384 * 600;
  float* w_y2   = ws + o; o += (size_t)16384 * 350;
  float* w_y3   = ws + o; o += (size_t)16384 * 200;
  float* w_cemb = ws + o; o += (size_t)16384 * 180;
  float* w_node = ws + o; o += (size_t)14 * 1024 * 180;
  float* w_ch   = ws + o; o += (size_t)1024 * 180;
  float* w_ncat = ws + o; o += (size_t)1024 * 395;
  float* w_t200 = ws + o; o += (size_t)1024 * 200;
  float* w_rh   = ws + o; o += (size_t)1024 * 180;
  float* w_r200 = ws + o; o += (size_t)1024 * 200;
  float* w_r180 = ws + o; o += (size_t)1024 * 180;

  const dim3 blk(256);
  const size_t sh_gemm = 16 * 32 * 4 + 128 * 32 * 2;                       // 10.0 KB
  const size_t sh_l100 = 400 * 32 * 2 + 400 * 128 * 2 + 400 * 4 +
                         16 * 32 * 4 + 16 * 128 * 2 + 16 * 400 * 4 + 2 * 16 * 100 * 4;  // ~170 KB
  const size_t sh_l180 = 16 * 192 * 4 + 16 * 192 * 2 + 16 * 720 * 4 + 2 * 16 * 180 * 4; // ~85.5 KB

  // 1) encode_vectors: [262144,8] @ [8,8]^T + b
  gemm_kernel<<<dim3(262144 / 16, 1), blk, sh_gemm, stream>>>(
      in_vecs, 8, encW, encB, w_vec, 8, 262144, 8, 8, 0);
  // 2) vec biLSTM -> prog [16384,200]
  lstm100_kernel<<<1024, blk, sh_l100, stream>>>(w_vec, 16, 8, 0, vfWi, vfWh, vfBi, vfBh, w_prog, 200, 0);
  lstm100_kernel<<<1024, blk, sh_l100, stream>>>(w_vec, 16, 8, 1, vbWi, vbWh, vbBi, vbBh, w_prog, 200, 100);
  // 3) expr LSTM -> [16384,100]
  lstm100_kernel<<<1024, blk, sh_l100, stream>>>(in_expr, 64, 11, 0, exprWi, exprWh, exprBi, exprBh, w_expr, 100, 0);
  // 4) concat + comp_mlp (ELU chain) -> comps_emb [16384,180]
  concat_comp_kernel<<<4096, blk, 0, stream>>>(in_first, w_prog, in_third, w_expr, w_xcat);
  gemm_kernel<<<dim3(1024, 5), blk, sh_gemm, stream>>>(w_xcat, 946, cmpW[0], cmpB[0], w_y1, 600, 16384, 600, 946, 1);
  gemm_kernel<<<dim3(1024, 3), blk, sh_gemm, stream>>>(w_y1, 600, cmpW[1], cmpB[1], w_y2, 350, 16384, 350, 600, 1);
  gemm_kernel<<<dim3(1024, 2), blk, sh_gemm, stream>>>(w_y2, 350, cmpW[2], cmpB[2], w_y3, 200, 16384, 200, 350, 1);
  gemm_kernel<<<dim3(1024, 2), blk, sh_gemm, stream>>>(w_y3, 200, cmpW[3], cmpB[3], w_cemb, 180, 16384, 180, 200, 1);

  // 5) tree aggregation, bottom-up (leaves 0-7, mids 8-11, tops 12-13)
  const int S = 1024 * 180;
  struct Nd { int loop, c0, c1, a, b; };
  const Nd tn[14] = {
      {3, 0, 1, -1, -1},  {4, 2, 3, -1, -1},   {5, 4, 5, -1, -1},   {6, 6, 7, -1, -1},
      {10, 8, 9, -1, -1}, {11, 10, 11, -1, -1},{12, 12, 13, -1, -1},{13, 14, 15, -1, -1},
      {1, -1, -1, 0, 1},  {2, -1, -1, 2, 3},   {8, -1, -1, 4, 5},   {9, -1, -1, 6, 7},
      {0, -1, -1, 8, 9},  {7, -1, -1, 10, 11}};
  for (int n = 0; n < 14; ++n) {
    if (tn[n].a < 0) {  // leaf: comps_lstm over comps_emb[:, [c0,c1], :]
      lstm180_seq2_kernel<<<64, blk, sh_l180, stream>>>(
          w_cemb, 16 * 180, 180, tn[n].c0, tn[n].c1, compsWi, compsWh, compsBi, compsBh, w_ch);
      node_concat_kernel<<<1580, blk, 0, stream>>>(w_ncat, noNodes, 1, w_ch, 0, in_loops, tn[n].loop);
    } else {            // internal: nodes_lstm over the two child embeddings
      lstm180_seq2_kernel<<<64, blk, sh_l180, stream>>>(
          w_node + (size_t)tn[n].a * S, 180, S, 0, tn[n].b - tn[n].a,
          nodesWi, nodesWh, nodesBi, nodesBh, w_ch);
      node_concat_kernel<<<1580, blk, 0, stream>>>(w_ncat, w_ch, 0, noComps, 1, in_loops, tn[n].loop);
    }
    gemm_kernel<<<dim3(64, 2), blk, sh_gemm, stream>>>(w_ncat, 395, catW0, catB0, w_t200, 200, 1024, 200, 395, 1);
    gemm_kernel<<<dim3(64, 2), blk, sh_gemm, stream>>>(w_t200, 200, catW1, catB1,
                                                       w_node + (size_t)n * S, 180, 1024, 180, 200, 1);
  }

  // 6) roots_lstm over the two root embeddings, reg_mlp, predict + LeakyReLU
  lstm180_seq2_kernel<<<64, blk, sh_l180, stream>>>(
      w_node + (size_t)12 * S, 180, S, 0, 1, rootsWi, rootsWh, rootsBi, rootsBh, w_rh);
  gemm_kernel<<<dim3(64, 2), blk, sh_gemm, stream>>>(w_rh, 180, regW0, regB0, w_r200, 200, 1024, 200, 180, 1);
  gemm_kernel<<<dim3(64, 2), blk, sh_gemm, stream>>>(w_r200, 200, regW1, regB1, w_r180, 180, 1024, 180, 200, 1);
  gemm_kernel<<<dim3(64, 1), blk, sh_gemm, stream>>>(w_r180, 180, predW, predB,
                                                     (float*)d_out, 1, 1024, 1, 180, 2);
}
